// Retention_32091995636018
// MI455X (gfx1250) — matrix operand order repaired and checked by reading, not compiled, not run on hardware
//
#include <hip/hip_runtime.h>

// Problem constants (B=2,H=8 flattened to BH=16)
#define BHN   16
#define LEN   2048
#define DMQ   256
#define VD    512
#define LOG2G      (-0.15200309344504997f)   // log2(0.9)
#define LN_G       (-0.10536051565782628f)   // ln(0.9)
#define INV1MG     (10.0f)                   // 1/(1-gamma)
#define L2_10000   (13.287712379549449f)     // log2(10000)

typedef __attribute__((ext_vector_type(16))) __bf16 v16bf;
typedef __attribute__((ext_vector_type(8)))  float  v8f;
typedef __attribute__((ext_vector_type(4)))  float  v4f;

union Frag {
  v16bf v;
  uint4 q[2];
};

// native RTNE f32 -> bf16 (single v_cvt on gfx1250)
static __device__ __forceinline__ unsigned short bfbits(float f) {
  __bf16 h = (__bf16)f;
  return __builtin_bit_cast(unsigned short, h);
}

// ---------------- kernel 0a: X fp32 -> bf16 ----------------
__global__ __launch_bounds__(256)
void cvt_x(const float* __restrict__ X, unsigned short* __restrict__ Xbf) {
  int i = blockIdx.x * blockDim.x + threadIdx.x;     // one float4 per thread
  float4 v = ((const float4*)X)[i];
  uint2 p;
  p.x = (unsigned)bfbits(v.x) | ((unsigned)bfbits(v.y) << 16);
  p.y = (unsigned)bfbits(v.z) | ((unsigned)bfbits(v.w) << 16);
  ((uint2*)Xbf)[i] = p;
}

// -------- kernel 0b: weights -> transposed packed bf16 Wt[1024][256] --------
__global__ __launch_bounds__(256)
void cvt_w(const float* __restrict__ Wq, const float* __restrict__ Wk,
           const float* __restrict__ Wv, unsigned short* __restrict__ Wt) {
  int i = blockIdx.x * blockDim.x + threadIdx.x;     // 0..262143
  int o = i >> 8;                                    // output column 0..1023
  int k = i & 255;
  float v;
  if (o < 256)        v = Wq[k * 256 + o];
  else if (o < 512)   v = Wk[k * 256 + (o - 256)];
  else                v = Wv[k * 512 + (o - 512)];
  Wt[i] = bfbits(v);
}

// ---------------- kernel 1: projections + xPos rotary ----------------
// 4 waves/block; each wave produces a 16(l) x 16(out) tile.
__global__ __launch_bounds__(128)
void proj_rope(const unsigned short* __restrict__ Xbf,
               const unsigned short* __restrict__ Wt,
               unsigned short* __restrict__ Qbf,
               unsigned short* __restrict__ Kbf,
               unsigned short* __restrict__ Vt) {
  const int lane = threadIdx.x & 31;
  const int w    = threadIdx.x >> 5;
  const int rowTile = blockIdx.x;              // 0..2047
  const int colTile = blockIdx.y * 4 + w;      // 0..63 (16 cols each)
  const int bh = rowTile >> 7;
  const int l0 = (rowTile & 127) << 4;
  const int hi = lane >> 4;
  const int lr = lane & 15;

  const uint4* X4 = (const uint4*)Xbf;
  const uint4* W4 = (const uint4*)Wt;
  const int arow = bh * LEN + l0 + lr;         // A row (row-major, stride 256)
  const int brow = colTile * 16 + lr;          // B row in Wt (stride 256)

  v8f acc = {};
  #pragma unroll
  for (int s = 0; s < 8; ++s) {
    Frag a, b;
    const int ka = 32 * s + 8 * hi;            // A frag: V0-3 / V4-7 K split
    a.q[0] = X4[(arow * DMQ + ka) >> 3];
    a.q[1] = X4[(arow * DMQ + ka + 16) >> 3];
    const int kb = 32 * s + 8 * hi;            // B frag: V0-3 / V4-7 K split
    b.q[0] = W4[(brow * DMQ + kb) >> 3];
    b.q[1] = W4[(brow * DMQ + kb + 16) >> 3];
    acc = __builtin_amdgcn_wmma_f32_16x16x32_bf16(false, a.v, false, b.v,
                                                  (short)0, acc, false, false);
  }

  const int n0g = colTile * 16;
  if (n0g < 512) {
    // Q or K with xPos rotary: pairs are adjacent columns -> lane^1 exchange
    const bool isK = (n0g >= 256);
    const int  c   = (isK ? n0g - 256 : n0g) + lr;        // 0..255
    const float invfreq = exp2f(-(float)(c >> 1) * (L2_10000 / 127.0f));
    const float sg = (((c & 1) == 0) == isK) ? 1.0f : -1.0f;
    unsigned short* dst = isK ? Kbf : Qbf;
    #pragma unroll
    for (int r = 0; r < 8; ++r) {
      const int l = l0 + r + 8 * hi;
      const float p  = __shfl_xor(acc[r], 1, 32);
      const float a  = (float)l * invfreq;
      const float sn = __sinf(a), cs = __cosf(a);
      dst[(bh * LEN + l) * DMQ + c] = bfbits(acc[r] * cs + sg * p * sn);
    }
  } else {
    // V: store transposed Vt[bh][vcol][l] so vt-GEMM B frags are contiguous
    const int vcol = n0g - 512 + lr;
    unsigned short pk[8];
    #pragma unroll
    for (int r = 0; r < 8; ++r) pk[r] = bfbits(acc[r]);
    uint4 u;
    u.x = (unsigned)pk[0] | ((unsigned)pk[1] << 16);
    u.y = (unsigned)pk[2] | ((unsigned)pk[3] << 16);
    u.z = (unsigned)pk[4] | ((unsigned)pk[5] << 16);
    u.w = (unsigned)pk[6] | ((unsigned)pk[7] << 16);
    ((uint4*)Vt)[(((bh * VD + vcol) * LEN) + l0 + 8 * hi) >> 3] = u;
  }
}

// ---------------- kernel 2: fused score + decay + vt ----------------
// 8 waves/block, one 16-row l-tile per block; wave w owns S-tile w of each
// 128-col chunk and a 64-wide VDIM slice of the vt accumulator.
__global__ __launch_bounds__(256)
void attention(const unsigned short* __restrict__ Qbf,
               const unsigned short* __restrict__ Kbf,
               const unsigned short* __restrict__ Vt,
               float* __restrict__ vtOut,
               float* __restrict__ scoreOut) {
  __shared__ __align__(16) __bf16 Sbuf[8 * 256]; // 8 x (16x16) bf16 tiles
  const int lane  = threadIdx.x & 31;
  const int w     = threadIdx.x >> 5;
  const int ltile = blockIdx.x;     // 0..127
  const int bh    = blockIdx.y;     // 0..15
  const int l0    = ltile << 4;
  const int lEnd  = l0 + 15;
  const int hi    = lane >> 4;
  const int lr    = lane & 15;

  const uint4* Q4 = (const uint4*)Qbf;
  const uint4* K4 = (const uint4*)Kbf;
  const uint4* V4 = (const uint4*)Vt;

  // Preload the 8 Q A-fragments for this row tile (stay in VGPRs all kernel)
  v16bf qf[8];
  {
    const int arow = bh * LEN + l0 + lr;
    #pragma unroll
    for (int s = 0; s < 8; ++s) {
      Frag a;
      const int ka = 32 * s + 8 * hi;
      a.q[0] = Q4[(arow * DMQ + ka) >> 3];
      a.q[1] = Q4[(arow * DMQ + ka + 16) >> 3];
      qf[s] = a.v;
    }
  }

  // Hoisted per-row decay scale: 1/(sqrt(s_l)*max(sqrt(s_l),1)) / sqrt(256)
  float fscale[8];
  #pragma unroll
  for (int r = 0; r < 8; ++r) {
    const int l = l0 + r + 8 * hi;
    const float s1 = (1.0f - exp2f((float)(l + 1) * LOG2G)) * INV1MG;
    const float sq = sqrtf(s1);
    fscale[r] = 0.0625f / (sq * fmaxf(sq, 1.0f));
  }

  v8f vacc[4] = {{}, {}, {}, {}};
  const int vcolBase = w * 64;
  const int cLast = lEnd >> 7;                 // last 128-col chunk with data

  for (int c = 0; c <= cLast; ++c) {
    const int m0w = c * 128 + w * 16;
    v8f sacc = {};
    if (m0w <= lEnd) {                         // wave-uniform branch
      const int brow = bh * LEN + m0w + lr;    // K rows as WMMA B operand
      #pragma unroll
      for (int s = 0; s < 8; ++s) {
        Frag b;
        const int kb = 32 * s + 8 * hi;
        b.q[0] = K4[(brow * DMQ + kb) >> 3];
        b.q[1] = K4[(brow * DMQ + kb + 16) >> 3];
        sacc = __builtin_amdgcn_wmma_f32_16x16x32_bf16(false, qf[s], false, b.v,
                                                       (short)0, sacc, false, false);
      }
      if (c < cLast)                            // hint next chunk's K rows
        __builtin_prefetch(&Kbf[(bh * LEN + (c + 1) * 128 + w * 16 + lr) * DMQ], 0, 1);
    }

    // decay scale, stream score out (NT), stage bf16 S tile in LDS
    const int m = m0w + lr;
    #pragma unroll
    for (int r = 0; r < 8; ++r) {
      const int l = l0 + r + 8 * hi;
      const int diff = l - m;
      const float f = (diff >= 0) ? __expf((float)diff * LN_G) * fscale[r] : 0.0f;
      const float sc = sacc[r] * f;
      __builtin_nontemporal_store(sc, &scoreOut[(size_t)(bh * LEN + l) * LEN + m]);
      Sbuf[w * 256 + (r + 8 * hi) * 16 + lr] = (__bf16)sc;
    }
    __syncthreads();

    // vt += S(16x128) @ V(128 x 64-slice); pack two 16-col S tiles per K=32 WMMA
    #pragma unroll
    for (int p = 0; p < 4; ++p) {
      Frag a;
      a.q[0] = *(const uint4*)&Sbuf[(2 * p)     * 256 + lr * 16 + 8 * hi];
      a.q[1] = *(const uint4*)&Sbuf[(2 * p + 1) * 256 + lr * 16 + 8 * hi];
      const int kelem = c * 128 + 32 * p + 8 * hi;     // l-positions in Vt row
      #pragma unroll
      for (int nt = 0; nt < 4; ++nt) {
        const int vcol = vcolBase + nt * 16 + lr;
        Frag b;
        b.q[0] = V4[((bh * VD + vcol) * LEN + kelem) >> 3];
        b.q[1] = V4[((bh * VD + vcol) * LEN + kelem + 16) >> 3];
        vacc[nt] = __builtin_amdgcn_wmma_f32_16x16x32_bf16(false, a.v, false, b.v,
                                                           (short)0, vacc[nt], false, false);
      }
    }
    __syncthreads();
  }

  // zero-fill strictly-above-diagonal score chunks with coalesced b128 NT stores
  {
    const int zr = threadIdx.x >> 4;           // 16 rows
    const int zc = (threadIdx.x & 15) * 8;     // 16 x 8-float segments = 128 cols
    const v4f z = {0.0f, 0.0f, 0.0f, 0.0f};
    for (int cz = cLast + 1; cz < 16; ++cz) {
      float* p = &scoreOut[(size_t)(bh * LEN + l0 + zr) * LEN + cz * 128 + zc];
      __builtin_nontemporal_store(z, (v4f*)p);
      __builtin_nontemporal_store(z, (v4f*)(p + 4));
    }
  }

  // write vt slice
  #pragma unroll
  for (int nt = 0; nt < 4; ++nt) {
    const int vcol = vcolBase + nt * 16 + lr;
    #pragma unroll
    for (int r = 0; r < 8; ++r) {
      const int l = l0 + r + 8 * hi;
      __builtin_nontemporal_store(vacc[nt][r],
                                  &vtOut[(size_t)(bh * LEN + l) * VD + vcol]);
    }
  }
}

extern "C" void kernel_launch(void* const* d_in, const int* in_sizes, int n_in,
                              void* d_out, int out_size, void* d_ws, size_t ws_size,
                              hipStream_t stream) {
  const float* X  = (const float*)d_in[0];
  const float* Wq = (const float*)d_in[1];
  const float* Wk = (const float*)d_in[2];
  const float* Wv = (const float*)d_in[3];
  float* out = (float*)d_out;

  // workspace layout (bf16 staging): Xbf | Wt | Qbf | Kbf | Vt  (~84.4 MB)
  char* ws = (char*)d_ws;
  unsigned short* Xbf = (unsigned short*)(ws);
  unsigned short* Wt  = (unsigned short*)(ws + 16777216);
  unsigned short* Qbf = (unsigned short*)(ws + 16777216 + 524288);
  unsigned short* Kbf = (unsigned short*)(ws + 16777216 + 524288 + 16777216);
  unsigned short* Vt  = (unsigned short*)(ws + 16777216 + 524288 + 2 * 16777216);

  float* vtOut    = out;                       // (B,H,L,VDIM) = 16,777,216 f32
  float* scoreOut = out + (size_t)16777216;    // (B,H,L,L)    = 67,108,864 f32

  cvt_x    <<<8192, 256, 0, stream>>>(X, Xbf);
  cvt_w    <<<1024, 256, 0, stream>>>(Wq, Wk, Wv, Wt);
  proj_rope<<<dim3(2048, 16), 128, 0, stream>>>(Xbf, Wt, Qbf, Kbf, Vt);
  attention<<<dim3(128, 16), 256, 0, stream>>>(Qbf, Kbf, Vt, vtOut, scoreOut);
}
